// WaveletEncoder_84353157694104
// MI455X (gfx1250) — matrix operand order repaired and checked by reading, not compiled, not run on hardware
//
#include <hip/hip_runtime.h>

typedef __attribute__((ext_vector_type(16))) _Float16 v16h;
typedef __attribute__((ext_vector_type(8)))  _Float16 v8h;
typedef __attribute__((ext_vector_type(8)))  float    v8f;
typedef __attribute__((ext_vector_type(4)))  int      v4i;

#define INV_SQRT2F 0.70710678118654752440f
#define LDS_AS __attribute__((address_space(3)))
#define GLB_AS __attribute__((address_space(1)))

union V16U { v16h v; v8h h[2]; };

// Load 8 consecutive f32 from global, convert to 8 f16.
static __device__ inline v8h cvt8(const float* __restrict__ s) {
  float4 a = *reinterpret_cast<const float4*>(s);
  float4 b = *reinterpret_cast<const float4*>(s + 4);
  v8h h;
  h[0] = (_Float16)a.x; h[1] = (_Float16)a.y; h[2] = (_Float16)a.z; h[3] = (_Float16)a.w;
  h[4] = (_Float16)b.x; h[5] = (_Float16)b.y; h[6] = (_Float16)b.z; h[7] = (_Float16)b.w;
  return h;
}

// Async global->LDS 16-byte copy (gfx1250 GLOBAL_LOAD_ASYNC_TO_LDS_B128,
// tracked by ASYNCcnt). Signature per clang-22 diagnostic:
// (v4i addrspace(1)* gsrc, v4i addrspace(3)* ldsdst, imm offset, imm cpol).
static __device__ inline void async_g2l_b128(const void* g, void* l) {
#if __has_builtin(__builtin_amdgcn_global_load_async_to_lds_b128)
  __builtin_amdgcn_global_load_async_to_lds_b128((GLB_AS v4i*)g, (LDS_AS v4i*)l, 0, 0);
#else
  *reinterpret_cast<v8h*>(l) = *reinterpret_cast<const v8h*>(g);
#endif
}

static __device__ inline void wait_async0() {
#if __has_builtin(__builtin_amdgcn_s_wait_asynccnt)
  __builtin_amdgcn_s_wait_asynccnt(0);
#else
  asm volatile("s_wait_asynccnt 0" ::: "memory");
#endif
}

// ---------------------------------------------------------------------------
// Kernel 1: build combined Haar x (antialiased bilinear-resize) weight table.
// G is [4][8][1024] f16, row index a = type*8 + j:
//   type 0: level-1 lo   (+1/sqrt2 per col pair,  resize 512->8, tri half-width 64)
//   type 1: level-1 hi   (+/-1/sqrt2 per col pair)
//   type 2: level-2 lolo (+1/2 per col quad,      resize 256->8, tri half-width 32)
//   type 3: level-2 hilo (+1/2,+1/2,-1/2,-1/2 per col quad)
// jax.image.resize bilinear (antialias=True): w[p] = max(0, 1-|sf-p|/ks),
// sf = (j+0.5)*ks - 0.5, ks = in/8, normalized per output pixel.
// ---------------------------------------------------------------------------
__global__ void wenc_prep(_Float16* __restrict__ G) {
  __shared__ float ssum[16];
  int tid = threadIdx.x;
  if (tid < 16) {
    int s = tid >> 3;                  // 0 -> 512-wide table, 1 -> 256-wide
    int j = tid & 7;
    int n = s ? 256 : 512;
    float inv = (float)(n >> 3);       // 64 or 32
    float sf = (j + 0.5f) * inv - 0.5f;
    float sum = 0.f;
    for (int p = 0; p < n; ++p)
      sum += fmaxf(1.f - fabsf(sf - (float)p) / inv, 0.f);
    ssum[tid] = sum;
  }
  __syncthreads();
  for (int e = tid; e < 4 * 8 * 1024; e += blockDim.x) {
    int t = e >> 13;
    int j = (e >> 10) & 7;
    int c = e & 1023;
    float val;
    if (t < 2) {
      int p = c >> 1;
      float inv = 64.f;
      float sf = (j + 0.5f) * inv - 0.5f;
      float raw = fmaxf(1.f - fabsf(sf - (float)p) / inv, 0.f);
      val = raw / ssum[j] * INV_SQRT2F;
      if (t == 1 && (c & 1)) val = -val;            // hi: even - odd
    } else {
      int p = c >> 2;
      float inv = 32.f;
      float sf = (j + 0.5f) * inv - 0.5f;
      float raw = fmaxf(1.f - fabsf(sf - (float)p) / inv, 0.f);
      val = raw / ssum[8 + j] * 0.5f;
      if (t == 3 && ((c & 3) >= 2)) val = -val;     // hi(level2 over lo1)
    }
    G[e] = (_Float16)val;
  }
}

// ---------------------------------------------------------------------------
// Kernel 2: U = x[b] (1024x1024) @ G^T (1024x32) via v_wmma_f32_16x16x32_f16.
// Streams the entire 256 MB input exactly once (HBM-bound, ~11 us floor).
// G (64 KB) is staged into LDS once per block via async global->LDS b128, so
// the hot loop's B-fragments are ds_load_b128 instead of global reloads.
// Grid: (8 row-groups, 64 images); 4 waves/block; 32 rows (2 M-tiles) per
// wave -> 4 WMMAs per B-pair per k-step.
// Stores U transposed as f16: UT[b][n(32)][r(1024)].
// ---------------------------------------------------------------------------
__global__ void __launch_bounds__(128) wenc_gemm1(const float* __restrict__ x,
                                                  const _Float16* __restrict__ G,
                                                  _Float16* __restrict__ UT) {
  __shared__ _Float16 sG[32 * 1024];   // 64 KB
  int tid  = threadIdx.x;
  int b    = blockIdx.y;
  int wave = tid >> 5;
  int lane = tid & 31;
  int hl   = lane >> 4;                 // lane-half select
  int l16  = lane & 15;
  int m0   = blockIdx.x * 128 + wave * 32;

  // Stage G into LDS: 4096 b128 transfers, 32 per thread, async.
#pragma unroll 4
  for (int i = tid * 8; i < 32 * 1024; i += 128 * 8)
    async_g2l_b128(G + i, sG + i);
  wait_async0();
  __syncthreads();

  const float* xr0 = x + ((size_t)b << 20) + (size_t)(m0 + l16) * 1024;
  const float* xr1 = xr0 + (size_t)16 * 1024;
  v8f acc00 = {0.f,0.f,0.f,0.f,0.f,0.f,0.f,0.f};
  v8f acc01 = acc00, acc10 = acc00, acc11 = acc00;

  for (int kc = 0; kc < 1024; kc += 32) {
    if (kc < 896) {
      __builtin_prefetch(xr0 + kc + 128, 0, 3);   // global_prefetch_b8
      __builtin_prefetch(xr1 + kc + 128, 0, 3);
    }
    // A-frag (ISA 16-bit A 16x32): lanes 0-15 hold K kc+0..7 / kc+16..23,
    // lanes 16-31 hold K kc+8..15 / kc+24..31; row = m + (lane&15).
    int a1 = kc + hl * 8;
    V16U A0, A1;
    A0.h[0] = cvt8(xr0 + a1);  A0.h[1] = cvt8(xr0 + a1 + 16);
    A1.h[0] = cvt8(xr1 + a1);  A1.h[1] = cvt8(xr1 + a1 + 16);
    // B-frag from LDS: column n = tile*16 + (lane&15);
    // lanes 0-15 hold K kc+0..7 / kc+16..23, lanes 16-31 hold K kc+8..15 / kc+24..31.
    int kb = kc + hl * 8;
    const _Float16* g0 = sG + (size_t)l16 * 1024 + kb;          // rows 0..15  (types 0,1)
    const _Float16* g1 = sG + (size_t)(16 + l16) * 1024 + kb;   // rows 16..31 (types 2,3)
    V16U B0, B1;
    B0.h[0] = *(const v8h*)g0;  B0.h[1] = *(const v8h*)(g0 + 16);
    B1.h[0] = *(const v8h*)g1;  B1.h[1] = *(const v8h*)(g1 + 16);

    acc00 = __builtin_amdgcn_wmma_f32_16x16x32_f16(false, A0.v, false, B0.v,
                                                   (short)0, acc00, false, false);
    acc01 = __builtin_amdgcn_wmma_f32_16x16x32_f16(false, A0.v, false, B1.v,
                                                   (short)0, acc01, false, false);
    acc10 = __builtin_amdgcn_wmma_f32_16x16x32_f16(false, A1.v, false, B0.v,
                                                   (short)0, acc10, false, false);
    acc11 = __builtin_amdgcn_wmma_f32_16x16x32_f16(false, A1.v, false, B1.v,
                                                   (short)0, acc11, false, false);
  }
  // C layout: element g -> row M = g + 8*(lane>=16), col N = lane&15.
  _Float16* ut = UT + ((size_t)b << 15);   // 32*1024 halves per image
#pragma unroll
  for (int g = 0; g < 8; ++g) {
    int r0 = m0 + g + hl * 8;
    int r1 = r0 + 16;
    ut[(size_t)l16 * 1024 + r0]        = (_Float16)acc00[g];
    ut[(size_t)(16 + l16) * 1024 + r0] = (_Float16)acc01[g];
    ut[(size_t)l16 * 1024 + r1]        = (_Float16)acc10[g];
    ut[(size_t)(16 + l16) * 1024 + r1] = (_Float16)acc11[g];
  }
}

// ---------------------------------------------------------------------------
// Kernel 3: V = G (32x1024) @ U (1024x32) per image (WMMA, 1 tile per wave),
// then feat lookup + 7->256 projection + bias. Grid: 64 blocks, 4 waves.
// ---------------------------------------------------------------------------
__global__ void __launch_bounds__(128) wenc_gemm2(const _Float16* __restrict__ G,
                                                  const _Float16* __restrict__ UT,
                                                  const float* __restrict__ W,
                                                  const float* __restrict__ bias,
                                                  float* __restrict__ out) {
  __shared__ float sV[32][32];
  int b    = blockIdx.x;
  int wave = threadIdx.x >> 5;
  int lane = threadIdx.x & 31;
  int hl   = lane >> 4;
  int l16  = lane & 15;
  int mt   = wave >> 1;                 // M-tile (rows of G)
  int nt   = wave & 1;                  // N-tile (cols of U)

  const _Float16* ga = G  + (size_t)(mt * 16 + l16) * 1024;
  const _Float16* ub = UT + ((size_t)b << 15) + (size_t)(nt * 16 + l16) * 1024;
  v8f acc = {0.f,0.f,0.f,0.f,0.f,0.f,0.f,0.f};

  for (int kc = 0; kc < 1024; kc += 32) {
    int a1 = kc + hl * 8;
    V16U A;
    A.h[0] = *(const v8h*)(ga + a1);
    A.h[1] = *(const v8h*)(ga + a1 + 16);
    int kb = kc + hl * 8;
    V16U B;
    B.h[0] = *(const v8h*)(ub + kb);
    B.h[1] = *(const v8h*)(ub + kb + 16);
    acc = __builtin_amdgcn_wmma_f32_16x16x32_f16(false, A.v, false, B.v,
                                                 (short)0, acc, false, false);
  }
#pragma unroll
  for (int g = 0; g < 8; ++g)
    sV[mt * 16 + g + hl * 8][nt * 16 + l16] = acc[g];
  __syncthreads();

  // wavedec2 map order: [cA2, cH2, cV2, cD2, cH1, cV1, cD1]
  // row/col type per map (0=L1lo, 1=L1hi, 2=L2lo, 3=L2hi):
  const int rowT[7] = {2, 3, 2, 3, 1, 0, 1};
  const int colT[7] = {2, 2, 3, 3, 0, 1, 1};

  int k = threadIdx.x >> 1;             // output position 0..63 (= i*8 + j)
  int i = k >> 3, j = k & 7;
  float feat[7];
#pragma unroll
  for (int m = 0; m < 7; ++m) feat[m] = sV[rowT[m] * 8 + i][colT[m] * 8 + j];

  int n0 = (threadIdx.x & 1) * 128;
  float* op = out + ((size_t)b << 14) + (size_t)k * 256;
  for (int nn = 0; nn < 128; ++nn) {
    int n = n0 + nn;
    float a = bias[n];
#pragma unroll
    for (int m = 0; m < 7; ++m) a += feat[m] * W[m * 256 + n];
    op[n] = a;
  }
}

// ---------------------------------------------------------------------------
extern "C" void kernel_launch(void* const* d_in, const int* in_sizes, int n_in,
                              void* d_out, int out_size, void* d_ws, size_t ws_size,
                              hipStream_t stream) {
  const float* x    = (const float*)d_in[0];   // [64,1,1024,1024]
  const float* W    = (const float*)d_in[1];   // [7,256]
  const float* bias = (const float*)d_in[2];   // [256]
  float* out = (float*)d_out;                  // [64,64,256]

  // Workspace: G (f16, 64 KB) | UT (f16, 64 images * 32 * 1024 = 4 MB)
  _Float16* G  = (_Float16*)d_ws;
  _Float16* UT = (_Float16*)((char*)d_ws + 65536);

  wenc_prep <<<1, 256, 0, stream>>>(G);
  wenc_gemm1<<<dim3(8, 64), 128, 0, stream>>>(x, G, UT);
  wenc_gemm2<<<64, 128, 0, stream>>>(G, UT, W, bias, out);
}
